// Attention_25769803776569
// MI455X (gfx1250) — compile-verified, matrix operand order checked by reading, not run on hardware
//
#include <hip/hip_runtime.h>
#include <math.h>

typedef _Float16 half16 __attribute__((ext_vector_type(16)));
typedef _Float16 half8  __attribute__((ext_vector_type(8)));
typedef _Float16 half4  __attribute__((ext_vector_type(4)));
typedef float    float8 __attribute__((ext_vector_type(8)));
typedef float    floatv4 __attribute__((ext_vector_type(4)));

#define N_TOKENS     4096
#define NUM_HEADS    16
#define NUM_KV_HEADS 4
#define HEAD_DIM     128
#define GQA          (NUM_HEADS / NUM_KV_HEADS)
#define BLOCK_M      128          // query rows per block (8 waves x 16 rows)
#define TILE_N       32           // keys per inner tile (WMMA K for P*V)
#define QSTR         (NUM_HEADS * HEAD_DIM)     // 2048
#define KSTR         (NUM_KV_HEADS * HEAD_DIM)  // 512

static __device__ __forceinline__ half16 cat8(half8 lo, half8 hi) {
  return __builtin_shufflevector(lo, hi, 0,1,2,3,4,5,6,7,8,9,10,11,12,13,14,15);
}

__launch_bounds__(256)
__global__ void fa_fwd_gqa_causal(const float* __restrict__ q,
                                  const float* __restrict__ k,
                                  const float* __restrict__ v,
                                  float* __restrict__ out) {
  __shared__ _Float16 ldsK[TILE_N][HEAD_DIM];   // 8 KB  [key][d]
  __shared__ _Float16 ldsVt[HEAD_DIM][TILE_N];  // 8 KB  [d][key] (transposed V)
  __shared__ _Float16 ldsP[8][16][TILE_N];      // 8 KB  per-wave P staging

  const int tid  = threadIdx.x;
  const int wv   = tid >> 5;
  const int lane = tid & 31;
  const int l16  = lane & 15;
  const int hi   = lane >> 4;

  const int qb  = blockIdx.x * BLOCK_M;
  const int qh  = blockIdx.y;          // query head
  const int hkv = qh / GQA;            // kv head
  const int q0  = qb + wv * 16;        // first query row owned by this wave

  // ---- Q fragments: A-layout (16x32 f16), lane l16 = row, 4 d-chunks ----
  half16 qa[4];
  {
    const float* qrow = q + (size_t)(q0 + l16) * QSTR + (size_t)qh * HEAD_DIM;
    #pragma unroll
    for (int dc = 0; dc < 4; ++dc) {
      half16 f;
      #pragma unroll
      for (int h = 0; h < 8; ++h)
        f[h] = (_Float16)qrow[dc * 32 + hi * 8 + h];
      #pragma unroll
      for (int h = 0; h < 8; ++h)
        f[h + 8] = (_Float16)qrow[dc * 32 + 16 + hi * 8 + h];
      qa[dc] = f;
    }
  }

  // ---- O accumulators (8 x 16x16 f32 tiles over d), online-softmax state ----
  float8 oacc[8];
  #pragma unroll
  for (int t = 0; t < 8; ++t) oacc[t] = (float8)0.f;
  float m_i[8], l_lane[8];              // l_lane: per-lane PARTIAL row sums
  #pragma unroll
  for (int r = 0; r < 8; ++r) { m_i[r] = -1e30f; l_lane[r] = 0.f; }

  // scale folded with log2(e) so we can use exp2 (v_exp_f32)
  const float qk_scale = 0.08838834764831845f * 1.4426950408889634f;

  const int n_tiles = (qb + BLOCK_M) / TILE_N;  // covers keys <= qb+127 (causal)
  for (int j = 0; j < n_tiles; ++j) {
    const int kbase = j * TILE_N;
    // ---- stage K/V tile to LDS (vectorized fp32 -> f16; V transposed) ----
    __syncthreads();
    #pragma unroll
    for (int pass = 0; pass < 4; ++pass) {
      const int idx4 = (pass * 256 + tid) * 4;         // element index in tile
      const int key  = idx4 >> 7;                      // / HEAD_DIM
      const int d    = idx4 & (HEAD_DIM - 1);
      const size_t g = (size_t)(kbase + key) * KSTR + (size_t)hkv * HEAD_DIM + d;
      const floatv4 kf = *(const floatv4*)(k + g);
      const floatv4 vf = *(const floatv4*)(v + g);
      half4 kh;
      #pragma unroll
      for (int i = 0; i < 4; ++i) kh[i] = (_Float16)kf[i];
      *(half4*)&ldsK[key][d] = kh;                     // ds_store_b64
      #pragma unroll
      for (int i = 0; i < 4; ++i) ldsVt[d + i][key] = (_Float16)vf[i];
    }
    // ---- prefetch next tile's K/V cachelines (global_prefetch_b8) ----
    if (j + 1 < n_tiles) {
      const int t2   = tid & 127;
      const int key  = t2 >> 1;
      const int dhalf = (t2 & 1) * 64;
      const float* base = (tid < 128) ? k : v;
      __builtin_prefetch(base + (size_t)(kbase + TILE_N + key) * KSTR +
                             (size_t)hkv * HEAD_DIM + dhalf, 0, 0);
    }
    __syncthreads();

    // waves whose rows are entirely before this key tile just hit barriers
    if (kbase <= q0 + 15) {
      // ---- S = Q * K^T : two 16x16 f32 accumulators (keys 0-15, 16-31) ----
      float8 s0 = (float8)0.f, s1 = (float8)0.f;
      #pragma unroll
      for (int dc = 0; dc < 4; ++dc) {
        const _Float16* kr0 = &ldsK[l16][dc * 32 + hi * 8];
        const _Float16* kr1 = &ldsK[l16 + 16][dc * 32 + hi * 8];
        half16 b0 = cat8(*(const half8*)kr0, *(const half8*)(kr0 + 16));
        half16 b1 = cat8(*(const half8*)kr1, *(const half8*)(kr1 + 16));
        s0 = __builtin_amdgcn_wmma_f32_16x16x32_f16(false, qa[dc], false, b0,
                                                    (short)0, s0, false, false);
        s1 = __builtin_amdgcn_wmma_f32_16x16x32_f16(false, qa[dc], false, b1,
                                                    (short)0, s1, false, false);
      }

      // ---- scale; causal mask only on diagonal-crossing tiles ----
      float sm0[8], sm1[8];
      if (kbase + (TILE_N - 1) > q0) {   // wave-uniform branch
        const int col0 = kbase + l16;
        const int col1 = col0 + 16;
        #pragma unroll
        for (int r = 0; r < 8; ++r) {
          const int row = q0 + r + 8 * hi;
          sm0[r] = (col0 <= row) ? s0[r] * qk_scale : -1e30f;
          sm1[r] = (col1 <= row) ? s1[r] * qk_scale : -1e30f;
        }
      } else {
        #pragma unroll
        for (int r = 0; r < 8; ++r) {
          sm0[r] = s0[r] * qk_scale;
          sm1[r] = s1[r] * qk_scale;
        }
      }

      // ---- online softmax: cross-lane max; row-sum kept per-lane ----
      float corr[8];
      #pragma unroll
      for (int r = 0; r < 8; ++r) {
        float mx = fmaxf(sm0[r], sm1[r]);
        mx = fmaxf(mx, __shfl_xor(mx, 1));
        mx = fmaxf(mx, __shfl_xor(mx, 2));
        mx = fmaxf(mx, __shfl_xor(mx, 4));
        mx = fmaxf(mx, __shfl_xor(mx, 8));
        const float mnew = fmaxf(m_i[r], mx);
        corr[r] = __builtin_exp2f(m_i[r] - mnew);
        m_i[r] = mnew;
        const float p0 = __builtin_exp2f(sm0[r] - mnew);
        const float p1 = __builtin_exp2f(sm1[r] - mnew);
        l_lane[r] = l_lane[r] * corr[r] + (p0 + p1);   // partial sum, no shfl
        // stage P (f16) into per-wave LDS region, C-layout coordinates
        ldsP[wv][r + 8 * hi][l16]      = (_Float16)p0;
        ldsP[wv][r + 8 * hi][l16 + 16] = (_Float16)p1;
      }

      // rescale O accumulators by per-row correction
      #pragma unroll
      for (int t = 0; t < 8; ++t)
        #pragma unroll
        for (int r = 0; r < 8; ++r) oacc[t][r] *= corr[r];

      // per-wave LDS is in-order; fence stores before the A-layout reload
      asm volatile("s_wait_dscnt 0" ::: "memory");

      // ---- reload P as A-fragment (16x32), then O += P * V ----
      const _Float16* pr = &ldsP[wv][l16][hi * 8];
      half16 pa = cat8(*(const half8*)pr, *(const half8*)(pr + 16));
      #pragma unroll
      for (int t = 0; t < 8; ++t) {
        const _Float16* vr = &ldsVt[t * 16 + l16][hi * 8];
        half16 bv = cat8(*(const half8*)vr, *(const half8*)(vr + 16));
        oacc[t] = __builtin_amdgcn_wmma_f32_16x16x32_f16(false, pa, false, bv,
                                                         (short)0, oacc[t],
                                                         false, false);
      }
    }
  }

  // ---- epilogue: reduce deferred row sums, normalize, store ----
  float l_inv[8];
  #pragma unroll
  for (int r = 0; r < 8; ++r) {
    float rs = l_lane[r];
    rs += __shfl_xor(rs, 1);
    rs += __shfl_xor(rs, 2);
    rs += __shfl_xor(rs, 4);
    rs += __shfl_xor(rs, 8);
    l_inv[r] = 1.0f / rs;
  }
  #pragma unroll
  for (int t = 0; t < 8; ++t) {
    #pragma unroll
    for (int r = 0; r < 8; ++r) {
      const int row = q0 + r + 8 * hi;
      out[(size_t)row * QSTR + (size_t)qh * HEAD_DIM + t * 16 + l16] =
          oacc[t][r] * l_inv[r];
    }
  }
}

extern "C" void kernel_launch(void* const* d_in, const int* in_sizes, int n_in,
                              void* d_out, int out_size, void* d_ws, size_t ws_size,
                              hipStream_t stream) {
  (void)in_sizes; (void)n_in; (void)out_size; (void)d_ws; (void)ws_size;
  const float* q = (const float*)d_in[0];
  const float* k = (const float*)d_in[1];
  const float* v = (const float*)d_in[2];
  float* out = (float*)d_out;
  dim3 grid(N_TOKENS / BLOCK_M, NUM_HEADS, 1);
  fa_fwd_gqa_causal<<<grid, 256, 0, stream>>>(q, k, v, out);
}
